// TimePointSelector_32976758899210
// MI455X (gfx1250) — matrix operand order repaired and checked by reading, not compiled, not run on hardware
//
#include <hip/hip_runtime.h>
#include <hip/hip_bf16.h>
#include <math.h>

#define HIDDEN 256
#define NPTS 1024
#define NROWS 64
#define KSEL 16

typedef __attribute__((ext_vector_type(16))) _Float16 v16h;
typedef __attribute__((ext_vector_type(8)))  float    v8f;

// ---------------------------------------------------------------------------
// Kernel 1: per-row min / 1/range  ->  stats[0..63]=tmin, stats[64..127]=invr
// ---------------------------------------------------------------------------
__global__ __launch_bounds__(256) void minmax_kernel(const float* __restrict__ tp,
                                                     float* __restrict__ stats) {
    __shared__ float smin[256];
    __shared__ float smax[256];
    const int r   = blockIdx.x;
    const int tid = threadIdx.x;
    const float* x = tp + r * NPTS;
    float vmn =  3.402823466e38f;
    float vmx = -3.402823466e38f;
    for (int i = tid; i < NPTS; i += 256) {
        float v = x[i];
        vmn = fminf(vmn, v);
        vmx = fmaxf(vmx, v);
    }
    smin[tid] = vmn; smax[tid] = vmx;
    __syncthreads();
    for (int off = 128; off > 0; off >>= 1) {
        if (tid < off) {
            smin[tid] = fminf(smin[tid], smin[tid + off]);
            smax[tid] = fmaxf(smax[tid], smax[tid + off]);
        }
        __syncthreads();
    }
    if (tid == 0) {
        float rng = fmaxf(smax[0] - smin[0], 1e-6f);
        stats[r]        = smin[0];
        stats[NROWS + r] = 1.0f / rng;
    }
}

// ---------------------------------------------------------------------------
// Kernel 2: fused MLP with WMMA.
//   Per block: 8 waves, each wave owns a 16-row M tile (block covers 128 rows).
//   K loop (256 in chunks of 32): stage W2^T chunk in LDS as f16, then each
//   wave does 16x v_wmma_f32_16x16x32_f16 (one per 16-col N tile).
//   Layer 3 (h2 @ W3 + b3, softplus) fused via shfl reduction.
// ---------------------------------------------------------------------------
__global__ __launch_bounds__(256) void mlp_kernel(
    const float* __restrict__ tp,
    const float* __restrict__ W1, const float* __restrict__ b1,
    const float* __restrict__ W2, const float* __restrict__ b2,
    const float* __restrict__ W3, const float* __restrict__ b3,
    const float* __restrict__ stats,
    float* __restrict__ density) {

    // W2^T chunk: sW2T[n*32 + kk] = (f16) W2[(kb*32+kk)*256 + n]   (16 KB)
    __shared__ _Float16 sW2T[HIDDEN * 32];

    const int lane  = threadIdx.x & 31;
    const int wave  = threadIdx.x >> 5;
    const int mrow0 = blockIdx.x * 128 + wave * 16;   // first row of wave's tile
    const int m     = lane & 15;
    const int grow  = mrow0 + m;                      // global flattened row
    const int batch = grow >> 10;                     // row / 1024

    const float t = (tp[grow] - stats[batch]) * stats[NROWS + batch];

    v8f acc[16];
#pragma unroll
    for (int i = 0; i < 16; ++i) acc[i] = (v8f){0.f,0.f,0.f,0.f,0.f,0.f,0.f,0.f};

    for (int kb = 0; kb < 8; ++kb) {
        __syncthreads();   // previous-iteration LDS reads done
        // cooperative staging, coalesced in n
        for (int idx = threadIdx.x; idx < HIDDEN * 32; idx += 256) {
            int n  = idx & 255;
            int kk = idx >> 8;
            sW2T[n * 32 + kk] = (_Float16)W2[(kb * 32 + kk) * HIDDEN + n];
        }
        __syncthreads();

        // A fragment: h1 = relu(t*W1 + b1), contiguous k = 16*(lane/16) + e, same order as B
        const int base0 = kb * 32 + ((lane < 16) ? 0 : 16);
        const int base1 = kb * 32 + ((lane < 16) ? 8 : 24);
        v16h a;
#pragma unroll
        for (int e = 0; e < 8; ++e) {
            int k0 = base0 + e, k1 = base1 + e;
            float h0 = t * W1[k0] + b1[k0]; h0 = h0 > 0.f ? h0 : 0.f;
            float h1 = t * W1[k1] + b1[k1]; h1 = h1 > 0.f ? h1 : 0.f;
            a[e]     = (_Float16)h0;
            a[e + 8] = (_Float16)h1;
        }

        // 16 N tiles of 16 columns each
#pragma unroll
        for (int nt = 0; nt < 16; ++nt) {
            const int n = nt * 16 + (lane & 15);
            const _Float16* src = &sW2T[n * 32 + ((lane < 16) ? 0 : 16)];
            v16h bfrag;
#pragma unroll
            for (int e = 0; e < 16; ++e) bfrag[e] = src[e];
            acc[nt] = __builtin_amdgcn_wmma_f32_16x16x32_f16(
                false, a, false, bfrag, (short)0, acc[nt], false, false);
        }
    }

    // Layer 3: density_pre[M] = sum_n relu(acc + b2[n]) * W3[n]
    // C layout: acc[nt] elem i -> (M = i + (lane<16 ? 0 : 8), N = nt*16 + (lane&15))
    float partial[8];
#pragma unroll
    for (int i = 0; i < 8; ++i) partial[i] = 0.f;
#pragma unroll
    for (int nt = 0; nt < 16; ++nt) {
        const int n  = nt * 16 + (lane & 15);
        const float b2n = b2[n];
        const float w3n = W3[n];
#pragma unroll
        for (int i = 0; i < 8; ++i) {
            float v = acc[nt][i] + b2n;
            v = v > 0.f ? v : 0.f;
            partial[i] += v * w3n;
        }
    }
    // reduce across the 16 lanes of each half-wave
#pragma unroll
    for (int off = 1; off < 16; off <<= 1) {
#pragma unroll
        for (int i = 0; i < 8; ++i)
            partial[i] += __shfl_xor(partial[i], off, 32);
    }
    if ((lane & 15) == 0) {
        const int rbase = mrow0 + ((lane < 16) ? 0 : 8);
        const float bb3 = b3[0];
#pragma unroll
        for (int i = 0; i < 8; ++i) {
            float z = partial[i] + bb3;
            float d = (z > 20.f) ? z : log1pf(expf(z));
            density[rbase + i] = d;
        }
    }
}

// ---------------------------------------------------------------------------
// Kernel 3: per-row peak prominence + top-16 selection (one block per row)
// ---------------------------------------------------------------------------
__global__ __launch_bounds__(256) void select_kernel(const float* __restrict__ density,
                                                     float* __restrict__ out_idx) {
    __shared__ float sx[NPTS];
    __shared__ float sc[NPTS];
    __shared__ float rv[256];
    __shared__ int   ri[256];
    __shared__ int   scount;
    __shared__ int   ssel[KSEL];

    const int r   = blockIdx.x;
    const int tid = threadIdx.x;
    const float* x = density + r * NPTS;

    if (tid == 0) scount = 0;
    for (int i = tid; i < NPTS; i += 256) sx[i] = x[i];
    __syncthreads();

    int localValid = 0;
    for (int i = tid; i < NPTS; i += 256) {
        const float xi = sx[i];
        const bool peak = (i > 0) && (i < NPTS - 1) && (xi > sx[i - 1]) && (xi > sx[i + 1]);
        // left scan: running min until a strictly-higher sample bounds us
        float lmin = xi;
        for (int j = i - 1; j >= 0; --j) {
            float v = sx[j];
            if (v > xi) break;
            lmin = fminf(lmin, v);
        }
        float rmin = xi;
        for (int j = i + 1; j < NPTS; ++j) {
            float v = sx[j];
            if (v > xi) break;
            rmin = fminf(rmin, v);
        }
        const float prom = xi - fmaxf(lmin, rmin);
        const bool valid = peak && (prom >= 0.1f);
        if (valid) localValid++;
        sc[i] = valid ? prom : -INFINITY;
    }
    if (localValid) atomicAdd(&scount, localValid);
    __syncthreads();

    const bool useProm = (scount >= KSEL);
    if (!useProm) {
        for (int i = tid; i < NPTS; i += 256) sc[i] = sx[i];
    }
    __syncthreads();

    // 16 rounds of tie-stable (lowest-index) argmax
    for (int k = 0; k < KSEL; ++k) {
        float bv = -INFINITY;
        int   bi = NPTS;
        for (int i = tid; i < NPTS; i += 256) {
            float s = sc[i];
            if (s > bv || (s == bv && i < bi)) { bv = s; bi = i; }
        }
        rv[tid] = bv; ri[tid] = bi;
        __syncthreads();
        for (int off = 128; off > 0; off >>= 1) {
            if (tid < off) {
                float ov = rv[tid + off];
                int   oi = ri[tid + off];
                if (ov > rv[tid] || (ov == rv[tid] && oi < ri[tid])) {
                    rv[tid] = ov; ri[tid] = oi;
                }
            }
            __syncthreads();
        }
        if (tid == 0) {
            ssel[k] = ri[0];
            sc[ri[0]] = -INFINITY;
        }
        __syncthreads();
    }

    if (tid == 0) {
        // insertion sort ascending
        for (int a = 1; a < KSEL; ++a) {
            int v = ssel[a];
            int b = a - 1;
            while (b >= 0 && ssel[b] > v) { ssel[b + 1] = ssel[b]; --b; }
            ssel[b + 1] = v;
        }
        for (int k = 0; k < KSEL; ++k)
            out_idx[r * KSEL + k] = (float)ssel[k];
    }
}

// ---------------------------------------------------------------------------
extern "C" void kernel_launch(void* const* d_in, const int* in_sizes, int n_in,
                              void* d_out, int out_size, void* d_ws, size_t ws_size,
                              hipStream_t stream) {
    (void)in_sizes; (void)n_in; (void)out_size; (void)ws_size;
    const float* tp = (const float*)d_in[0];
    // d_in[1] = num_gen_points (scalar int) -> fixed K=16
    const float* W1 = (const float*)d_in[2];
    const float* b1 = (const float*)d_in[3];
    const float* W2 = (const float*)d_in[4];
    const float* b2 = (const float*)d_in[5];
    const float* W3 = (const float*)d_in[6];
    const float* b3 = (const float*)d_in[7];

    float* density = (float*)d_out;                       // 64*1024
    float* out_idx = (float*)d_out + NROWS * NPTS;        // 64*16 (as float)
    float* stats   = (float*)d_ws;                        // tmin[64], invr[64]

    minmax_kernel<<<NROWS, 256, 0, stream>>>(tp, stats);
    mlp_kernel<<<(NROWS * NPTS) / 128, 256, 0, stream>>>(
        tp, W1, b1, W2, b2, W3, b3, stats, density);
    select_kernel<<<NROWS, 256, 0, stream>>>(density, out_idx);
}
